// SelfExpression2_7430293422438
// MI455X (gfx1250) — matrix operand order repaired and checked by reading, not compiled, not run on hardware
//
#include <hip/hip_runtime.h>
#include <hip/hip_bf16.h>

// ---------------------------------------------------------------------------
// y = C2 @ adj @ x  reassociated as  C2 @ (adj @ x), adj never materialized.
//   adj @ x = dinv ⊙ (Asym @ (dinv⊙x)) + dinv ⊙ (dinv⊙x),  Asym = (A+A^T)/2
// Both big GEMMs run on v_wmma_f32_16x16x32_bf16 (fp32 accumulate).
// ---------------------------------------------------------------------------

#define NROWS 8192
#define DCOLS 1024

typedef __attribute__((ext_vector_type(16))) __bf16 v16bf;
typedef __attribute__((ext_vector_type(8)))  __bf16 v8bf;
typedef __attribute__((ext_vector_type(4)))  __bf16 v4bf;
typedef __attribute__((ext_vector_type(8)))  float  v8f;

// ---------------- prep: row sums ----------------
__global__ __launch_bounds__(256) void rowsum_kernel(const float* __restrict__ A,
                                                     float* __restrict__ r) {
  __shared__ float red[256];
  const int row = blockIdx.x;
  const float* ap = A + (size_t)row * NROWS;
  float s = 0.f;
  for (int j = threadIdx.x; j < NROWS; j += 256) s += ap[j];
  red[threadIdx.x] = s;
  __syncthreads();
  for (int off = 128; off > 0; off >>= 1) {
    if (threadIdx.x < off) red[threadIdx.x] += red[threadIdx.x + off];
    __syncthreads();
  }
  if (threadIdx.x == 0) r[row] = red[0];
}

// ---------------- prep: column sums (fully coalesced) ----------------
__global__ __launch_bounds__(256) void colsum_kernel(const float* __restrict__ A,
                                                     float* __restrict__ c) {
  const int col = blockIdx.x * 256 + threadIdx.x;
  float s0 = 0.f, s1 = 0.f, s2 = 0.f, s3 = 0.f;
  for (int i = 0; i < NROWS; i += 4) {
    s0 += A[(size_t)(i + 0) * NROWS + col];
    s1 += A[(size_t)(i + 1) * NROWS + col];
    s2 += A[(size_t)(i + 2) * NROWS + col];
    s3 += A[(size_t)(i + 3) * NROWS + col];
  }
  c[col] = (s0 + s1) + (s2 + s3);
}

// ---------------- prep: dinv = ((r+c)/2 + 1)^{-1/2} ----------------
__global__ __launch_bounds__(256) void dinv_kernel(const float* __restrict__ r,
                                                   const float* __restrict__ c,
                                                   float* __restrict__ dinv) {
  const int i = blockIdx.x * 256 + threadIdx.x;
  const float deg = 0.5f * (r[i] + c[i]) + 1.0f;
  dinv[i] = (deg > 0.f) ? rsqrtf(deg) : 0.f;
}

// ---------------- prep: xsT[d][i] = bf16(dinv[i] * x[i][d]) (LDS transpose) ----
__global__ __launch_bounds__(256) void xsT_kernel(const float* __restrict__ x,
                                                  const float* __restrict__ dinv,
                                                  __bf16* __restrict__ xsT) {
  __shared__ float t[32][33];
  const int i0 = blockIdx.x * 32;
  const int d0 = blockIdx.y * 32;
  const int tx = threadIdx.x, ty = threadIdx.y;
#pragma unroll
  for (int rr = 0; rr < 4; ++rr) {
    const int i = i0 + ty + rr * 8;
    t[ty + rr * 8][tx] = x[(size_t)i * DCOLS + d0 + tx];
  }
  __syncthreads();
  const int ii = i0 + tx;
  const float dv = dinv[ii];
#pragma unroll
  for (int rr = 0; rr < 4; ++rr) {
    const int d = d0 + ty + rr * 8;
    xsT[(size_t)d * NROWS + ii] = (__bf16)(dv * t[tx][ty + rr * 8]);
  }
}

// ---------------- WMMA GEMM ----------------
// MODE 1: acc = Asym @ xs; A-tile symmetrized fp32->bf16 at staging via LDS
//         scratch; epilogue writes MT[n][m] = bf16(dinv[m]*(acc + xs[m][n]))
// MODE 0: acc = C2(fp32->bf16 at staging) @ M; epilogue writes fp32 out[m][n]
// Block tile 128x256, wave grid 2x4 (wave tile 64x64 = 16 WMMA tiles, 128 acc VGPRs).
template <int MODE>
__global__ __launch_bounds__(256) void gemm_kernel(
    const float* __restrict__ Amat,   // [8192][8192] fp32 (A or C2)
    const __bf16* __restrict__ BT,    // [1024][8192] bf16, B transposed
    const float* __restrict__ dinv,   // MODE1 only
    const __bf16* __restrict__ xsT,   // MODE1 only (additive term, transposed)
    float* __restrict__ outF,         // MODE0: [8192][1024]
    __bf16* __restrict__ outT)        // MODE1: [1024][8192]
{
  constexpr int K = 8192;
  constexpr int BM = 128, BN = 256, BK = 32;
  constexpr int LDA_B = BK + 8;        // 40 bf16 (80B rows, 16B aligned)
  constexpr int LDT   = BM + 4;        // 132 floats (528B rows, 16B aligned)
  constexpr int ABYTES = BM * LDA_B * 2;            // 10240 B  bf16 operand tile
  constexpr int TBYTES = BK * LDT * 4;              // 16896 B  fp32 A^T scratch

  __shared__ __align__(16) unsigned char smem[MODE ? (ABYTES + TBYTES) : ABYTES];
  __bf16* AsB = reinterpret_cast<__bf16*>(smem);
  float*  T2  = reinterpret_cast<float*>(smem + ABYTES);  // [BK][LDT] (MODE1)

  const int tid  = threadIdx.x;
  const int lane = tid & 31;
  const int wave = tid >> 5;
  const int wm = wave >> 2;  // 0..1
  const int wn = wave & 3;   // 0..3
  const int m0 = blockIdx.x * BM;
  const int n0 = blockIdx.y * BN;
  const int l16 = lane & 15;
  const int lh  = lane >> 4;  // 0/1

  v8f acc[4][4];
#pragma unroll
  for (int i = 0; i < 4; ++i)
#pragma unroll
    for (int j = 0; j < 4; ++j)
      acc[i][j] = v8f{0.f, 0.f, 0.f, 0.f, 0.f, 0.f, 0.f, 0.f};

  for (int k0 = 0; k0 < K; k0 += BK) {
    __syncthreads();  // protect previous iteration's fragment reads
    if (MODE == 1) {
      // pass 1: T2[kk][m] = A[k0+kk][m0+m]   (coalesced global + b128 LDS store)
#pragma unroll
      for (int u = 0; u < 4; ++u) {
        const int q  = tid + 256 * u;       // 0..1023
        const int kk = q >> 5;              // 0..31
        const int ms = (q & 31) << 2;       // 0..124
        const float4 v = *reinterpret_cast<const float4*>(
            Amat + (size_t)(k0 + kk) * K + (m0 + ms));
        *reinterpret_cast<float4*>(&T2[kk * LDT + ms]) = v;
      }
      __syncthreads();
      // pass 2: AsB[r][k] = bf16(0.5*(A[m0+r][k0+k] + T2[k][r]))
#pragma unroll
      for (int u = 0; u < 4; ++u) {
        const int q  = tid + 256 * u;
        const int r  = q >> 3;              // 0..127
        const int ks = (q & 7) << 2;        // 0..28
        const float4 a = *reinterpret_cast<const float4*>(
            Amat + (size_t)(m0 + r) * K + (k0 + ks));
        const float t0 = T2[(ks + 0) * LDT + r];
        const float t1 = T2[(ks + 1) * LDT + r];
        const float t2 = T2[(ks + 2) * LDT + r];
        const float t3 = T2[(ks + 3) * LDT + r];
        v4bf b;
        b[0] = (__bf16)(0.5f * (a.x + t0));
        b[1] = (__bf16)(0.5f * (a.y + t1));
        b[2] = (__bf16)(0.5f * (a.z + t2));
        b[3] = (__bf16)(0.5f * (a.w + t3));
        *reinterpret_cast<v4bf*>(&AsB[r * LDA_B + ks]) = b;
        if (k0 + BK < K)
          __builtin_prefetch(Amat + (size_t)(m0 + r) * K + (k0 + BK + ks), 0, 3);
      }
    } else {
      // plain fp32 -> bf16 conversion at staging time
#pragma unroll
      for (int u = 0; u < 4; ++u) {
        const int q  = tid + 256 * u;
        const int r  = q >> 3;
        const int ks = (q & 7) << 2;
        const float4 v = *reinterpret_cast<const float4*>(
            Amat + (size_t)(m0 + r) * K + (k0 + ks));
        v4bf b;
        b[0] = (__bf16)v.x; b[1] = (__bf16)v.y;
        b[2] = (__bf16)v.z; b[3] = (__bf16)v.w;
        *reinterpret_cast<v4bf*>(&AsB[r * LDA_B + ks]) = b;
        if (k0 + BK < K)
          __builtin_prefetch(Amat + (size_t)(m0 + r) * K + (k0 + BK + ks), 0, 3);
      }
    }
    __syncthreads();

    // ---- fragments (bf16 LDS tile, identical for both modes) ----
    // A 16x32 bf16 layout: lane l (m = l&15): K = {koff..koff+7, koff+16..koff+23},
    // koff = (l>>4)*8; elements packed 2-per-VGPR in K order.
    v16bf afrag[4], bfrag[4];
    const int koff = lh * 8;
#pragma unroll
    for (int i = 0; i < 4; ++i) {
      const int mr = wm * 64 + i * 16 + l16;
      const __bf16* rp = &AsB[mr * LDA_B];
      const v8bf lo = *reinterpret_cast<const v8bf*>(rp + koff);
      const v8bf hi = *reinterpret_cast<const v8bf*>(rp + koff + 16);
      afrag[i] = __builtin_shufflevector(lo, hi, 0, 1, 2, 3, 4, 5, 6, 7,
                                         8, 9, 10, 11, 12, 13, 14, 15);
    }
    // B 32x16 bf16 layout: lane l: col = l&15, K = {kb..kb+7, kb+16..kb+23} ->
    // two 16B reads from transposed BT row (L2-resident, 16 MB).
    const int kb = k0 + lh * 8;
#pragma unroll
    for (int j = 0; j < 4; ++j) {
      const int nr = n0 + wn * 64 + j * 16 + l16;
      const __bf16* bp = BT + (size_t)nr * K + kb;
      const v8bf lo = *reinterpret_cast<const v8bf*>(bp);
      const v8bf hi = *reinterpret_cast<const v8bf*>(bp + 16);
      bfrag[j] = __builtin_shufflevector(lo, hi, 0, 1, 2, 3, 4, 5, 6, 7,
                                         8, 9, 10, 11, 12, 13, 14, 15);
    }

#pragma unroll
    for (int i = 0; i < 4; ++i)
#pragma unroll
      for (int j = 0; j < 4; ++j)
        acc[i][j] = __builtin_amdgcn_wmma_f32_16x16x32_bf16(
            false, afrag[i], false, bfrag[j], (short)0, acc[i][j], false, false);
  }

  // ---- epilogue ----
  // D layout: lane l: n = l&15 (lane>=16 selects M+8), VGPR v -> M = base + 8*lh + v.
#pragma unroll
  for (int i = 0; i < 4; ++i) {
    const int mt = m0 + wm * 64 + i * 16 + 8 * lh;  // 8 consecutive rows per lane
    float dv[8];
    if (MODE == 1) {
      const float4 d0 = *reinterpret_cast<const float4*>(dinv + mt);
      const float4 d1 = *reinterpret_cast<const float4*>(dinv + mt + 4);
      dv[0] = d0.x; dv[1] = d0.y; dv[2] = d0.z; dv[3] = d0.w;
      dv[4] = d1.x; dv[5] = d1.y; dv[6] = d1.z; dv[7] = d1.w;
    }
#pragma unroll
    for (int j = 0; j < 4; ++j) {
      const int n = n0 + wn * 64 + j * 16 + l16;
      if (MODE == 1) {
        // M[mt+v][n] = dinv*(acc + xs); stored transposed (contiguous 16B/lane)
        const v8bf xv = *reinterpret_cast<const v8bf*>(xsT + (size_t)n * K + mt);
        v8bf o;
#pragma unroll
        for (int v = 0; v < 8; ++v)
          o[v] = (__bf16)(dv[v] * (acc[i][j][v] + (float)xv[v]));
        *reinterpret_cast<v8bf*>(outT + (size_t)n * K + mt) = o;
      } else {
#pragma unroll
        for (int v = 0; v < 8; ++v)
          outF[(size_t)(mt + v) * DCOLS + n] = acc[i][j][v];
      }
    }
  }
}

// ---------------------------------------------------------------------------
extern "C" void kernel_launch(void* const* d_in, const int* in_sizes, int n_in,
                              void* d_out, int out_size, void* d_ws, size_t ws_size,
                              hipStream_t stream) {
  (void)in_sizes; (void)n_in; (void)out_size; (void)ws_size;
  const float* x  = (const float*)d_in[0];  // [8192][1024]
  const float* C2 = (const float*)d_in[1];  // [8192][8192]
  const float* A  = (const float*)d_in[2];  // [8192][8192]
  float* out = (float*)d_out;               // [8192][1024]

  char* ws = (char*)d_ws;
  float*  r    = (float*)(ws);                       // 32 KB
  float*  c    = (float*)(ws + (32 << 10));          // 32 KB
  float*  dinv = (float*)(ws + (64 << 10));          // 32 KB
  __bf16* xsT  = (__bf16*)(ws + (128 << 10));        // 16 MB: [1024][8192]
  __bf16* MT   = (__bf16*)(ws + (128 << 10) + (size_t)DCOLS * NROWS * 2);  // 16 MB

  rowsum_kernel<<<NROWS, 256, 0, stream>>>(A, r);
  colsum_kernel<<<NROWS / 256, 256, 0, stream>>>(A, c);
  dinv_kernel<<<NROWS / 256, 256, 0, stream>>>(r, c, dinv);
  xsT_kernel<<<dim3(NROWS / 32, DCOLS / 32), dim3(32, 8), 0, stream>>>(x, dinv, xsT);
  // GEMM1: MT = transpose( dinv ⊙ (Asym @ xs + xs) )   [bf16]
  gemm_kernel<1><<<dim3(NROWS / 128, DCOLS / 256), 256, 0, stream>>>(
      A, xsT, dinv, xsT, nullptr, MT);
  // GEMM2: out = C2 @ M   [fp32 out]
  gemm_kernel<0><<<dim3(NROWS / 128, DCOLS / 256), 256, 0, stream>>>(
      C2, MT, nullptr, nullptr, out, nullptr);
}
